// EquivariantUpdate_79267916415020
// MI455X (gfx1250) — compile-verified, matrix operand order checked by reading, not run on hardware
//
#include <hip/hip_runtime.h>

// ---------------------------------------------------------------------------
// EGNN coordinate-update (edge MLP 257->128->128->1 + segment scatter) for
// MI455X / gfx1250.  Matrix work on v_wmma_f32_16x16x32_bf16; weights staged
// in LDS transposed+K-pair-packed so every WMMA fragment loads as 2x
// ds_load_b128; all 8 B fragments of a K-step preloaded so DS latency
// overlaps the XDL pipe; A fragments double-buffered across K-steps to avoid
// WMMA->VALU WAR nops; bf16 packing uses native v_cvt_pk_bf16_f32.
// ---------------------------------------------------------------------------

typedef __attribute__((ext_vector_type(16))) __bf16        v16bf;
typedef __attribute__((ext_vector_type(2)))  __bf16        v2bf;
typedef __attribute__((ext_vector_type(8)))  float         v8f;
typedef __attribute__((ext_vector_type(4)))  float         v4f;
typedef __attribute__((ext_vector_type(4)))  unsigned int  v4u;

#define NNODES 50000
#define NEDGES 800000
#define HID    128
#define EDGES_PER_WG 128
#define THREADS 256

// LDS strides (elements). Rows 16B-aligned so fragments load as ds_load_b128.
#define W1_STRIDE 132   // dwords per n-row: 128 K-pairs + 4 pad
#define W2_STRIDE 68    // dwords per n-row:  64 K-pairs + 4 pad
#define X1_STRIDE 136   // ushorts per edge-row: 128 feats + 8 pad (68 dwords)

#define SMEM_BYTES (128*W1_STRIDE*4 + 128*W2_STRIDE*4 + 128*X1_STRIDE*2 + 6*128*4)

__device__ __forceinline__ unsigned int pack2(float lo, float hi) {
  v2bf p;
  p[0] = (__bf16)lo;            // lowers to v_cvt_pk_bf16_f32
  p[1] = (__bf16)hi;
  return __builtin_bit_cast(unsigned int, p);
}
__device__ __forceinline__ float silu_f(float x) {
  return x / (1.0f + __expf(-x));
}

union Frag {
  v16bf v;
  unsigned int u[8];
  v4u q[2];
};

__device__ __forceinline__ void load_frag_lds(Frag& f, const unsigned int* rp) {
  f.q[0] = *(const v4u*)rp;           // kp = base + {0..3}
  f.q[1] = *(const v4u*)(rp + 8);     // kp = base + 8 + {0..3}
}

__global__ __launch_bounds__(THREADS)
void egnn_edge_kernel(const float* __restrict__ h,
                      const int*   __restrict__ edge_index,
                      const float* __restrict__ coord_diff,
                      const float* __restrict__ edge_attr,
                      const float* __restrict__ edge_mask,
                      const float* __restrict__ W1,
                      const float* __restrict__ b1,
                      const float* __restrict__ W2,
                      const float* __restrict__ b2,
                      const float* __restrict__ W3,
                      float*       __restrict__ out)
{
  extern __shared__ unsigned char smem[];
  unsigned int*   w1t = (unsigned int*)smem;                 // [n=128][W1_STRIDE]
  unsigned int*   w2t = w1t + 128 * W1_STRIDE;               // [n=128][W2_STRIDE]
  unsigned short* x1t = (unsigned short*)(w2t + 128 * W2_STRIDE); // [128][X1_STRIDE]
  float* w1r  = (float*)(x1t + 128 * X1_STRIDE);             // W1 row 256
  float* b1s  = w1r + 128;
  float* b2s  = b1s + 128;
  float* w3s  = b2s + 128;
  float* eas  = w3s + 128;                                   // edge_attr tile
  float* phis = eas + 128;                                   // phi per edge

  const int tid  = threadIdx.x;
  const int e0wg = blockIdx.x * EDGES_PER_WG;

  // ---- cooperative weight staging: f32 -> bf16, transposed, K-pair packed --
  // w1t[n][kp] = {bf16(W1[2kp][n]), bf16(W1[2kp+1][n])}
  for (int idx = tid; idx < 128 * 128; idx += THREADS) {
    int n = idx & 127, kp = idx >> 7;             // consecutive n -> coalesced
    w1t[n * W1_STRIDE + kp] =
        pack2(W1[(2 * kp) * HID + n], W1[(2 * kp + 1) * HID + n]);
  }
  for (int idx = tid; idx < 64 * 128; idx += THREADS) {
    int n = idx & 127, kp = idx >> 7;
    w2t[n * W2_STRIDE + kp] =
        pack2(W2[(2 * kp) * HID + n], W2[(2 * kp + 1) * HID + n]);
  }
  if (tid < 128) {
    w1r[tid] = W1[256 * HID + tid];
    b1s[tid] = b1[tid];
    b2s[tid] = b2[tid];
    w3s[tid] = W3[tid];
    eas[tid] = edge_attr[e0wg + tid];
  }
  __syncthreads();

  // ---- per-wave tile of 16 edges ------------------------------------------
  const int wave = tid >> 5;          // 0..7 -> edge tile within WG
  const int lane = tid & 31;
  const int half = lane >> 4;         // K-half selector (A/B layout)
  const int lm   = lane & 15;         // M (for A) / N (for B) lane index
  const int e0   = e0wg + wave * 16;

  const int erow = edge_index[e0 + lm];
  const int ecol = edge_index[NEDGES + e0 + lm];
  const float* hrow = h + (long)erow * HID;
  const float* hcol = h + (long)ecol * HID;

  // ================= Layer 1: [16 x 257] @ [257 x 128] =====================
  v8f acc[8];
#pragma unroll
  for (int t = 0; t < 8; ++t)
#pragma unroll
    for (int i = 0; i < 8; ++i) acc[t][i] = 0.0f;

  Frag Aab[2];   // A double buffer: kstep parity -> no WAR on in-flight WMMA

#pragma unroll
  for (int ks = 0; ks < 8; ++ks) {
    const float* src = (ks < 4) ? hrow : hcol;
    const int kbase = (ks & 3) * 32;

    // A fragment: two runs of 8 consecutive floats -> 4x global_load_b128
    Frag& A = Aab[ks & 1];
    {
      const v4f* p0 = (const v4f*)(src + kbase + half * 8);
      const v4f* p1 = (const v4f*)(src + kbase + 16 + half * 8);
      v4f f0 = p0[0], f1 = p0[1], f2 = p1[0], f3 = p1[1];
      A.u[0] = pack2(f0[0], f0[1]);  A.u[1] = pack2(f0[2], f0[3]);
      A.u[2] = pack2(f1[0], f1[1]);  A.u[3] = pack2(f1[2], f1[3]);
      A.u[4] = pack2(f2[0], f2[1]);  A.u[5] = pack2(f2[2], f2[3]);
      A.u[6] = pack2(f3[0], f3[1]);  A.u[7] = pack2(f3[2], f3[3]);
    }

    // Preload all 8 B fragments (16x ds_load_b128), then drain with WMMAs.
    const unsigned int* wbase = w1t + lm * W1_STRIDE + ks * 16 + half * 4;
    Frag B[8];
#pragma unroll
    for (int nt = 0; nt < 8; ++nt)
      load_frag_lds(B[nt], wbase + nt * 16 * W1_STRIDE);
#pragma unroll
    for (int nt = 0; nt < 8; ++nt)
      acc[nt] = __builtin_amdgcn_wmma_f32_16x16x32_bf16(
          false, A.v, false, B[nt].v, (short)0, acc[nt], false, false);
  }

  // epilogue: + b1 + edge_attr * W1[256,:], SiLU, stage as bf16 A for layer 2
#pragma unroll
  for (int nt = 0; nt < 8; ++nt) {
    const int n  = nt * 16 + lm;
    const float bn = b1s[n];
    const float wn = w1r[n];
#pragma unroll
    for (int r = 0; r < 8; ++r) {
      const int ml = r + half * 8;                 // C layout: M = r + half*8
      float val = acc[nt][r] + bn + eas[wave * 16 + ml] * wn;
      val = silu_f(val);
      x1t[(wave * 16 + ml) * X1_STRIDE + n] =
          __builtin_bit_cast(unsigned short, (__bf16)val);
    }
  }
  __syncthreads();

  // ================= Layer 2: [16 x 128] @ [128 x 128] =====================
  const unsigned int* x1u = (const unsigned int*)x1t;  // 68 dwords per edge
  v8f acc2[8];
#pragma unroll
  for (int t = 0; t < 8; ++t)
#pragma unroll
    for (int i = 0; i < 8; ++i) acc2[t][i] = 0.0f;

#pragma unroll
  for (int ks = 0; ks < 4; ++ks) {
    Frag& A = Aab[ks & 1];
    load_frag_lds(A, x1u + (wave * 16 + lm) * (X1_STRIDE / 2) + ks * 16 + half * 4);

    const unsigned int* wbase = w2t + lm * W2_STRIDE + ks * 16 + half * 4;
    Frag B[8];
#pragma unroll
    for (int nt = 0; nt < 8; ++nt)
      load_frag_lds(B[nt], wbase + nt * 16 * W2_STRIDE);
#pragma unroll
    for (int nt = 0; nt < 8; ++nt)
      acc2[nt] = __builtin_amdgcn_wmma_f32_16x16x32_bf16(
          false, A.v, false, B[nt].v, (short)0, acc2[nt], false, false);
  }

  // ===== Layer 3: SiLU then dot with W3 (128 -> 1), butterfly reduce =======
  float part[8];
#pragma unroll
  for (int r = 0; r < 8; ++r) part[r] = 0.0f;
#pragma unroll
  for (int nt = 0; nt < 8; ++nt) {
    const int n = nt * 16 + lm;
    const float bn = b2s[n];
    const float wn = w3s[n];
#pragma unroll
    for (int r = 0; r < 8; ++r)
      part[r] += silu_f(acc2[nt][r] + bn) * wn;
  }
#pragma unroll
  for (int r = 0; r < 8; ++r) {
    float p = part[r];
    p += __shfl_xor(p, 1, 32);
    p += __shfl_xor(p, 2, 32);
    p += __shfl_xor(p, 4, 32);
    p += __shfl_xor(p, 8, 32);        // all 16 lanes of the half hold the sum
    if (lm == 0) phis[wave * 16 + half * 8 + r] = p;
  }
  __syncthreads();

  // ===== scatter: out[row] += coord_diff * phi * edge_mask / 100 ===========
  if (lane < 16) {
    const int e = e0 + lane;
    const float scale = phis[wave * 16 + lane] * edge_mask[e] * 0.01f;
#pragma unroll
    for (int d = 0; d < 3; ++d) {
      atomicAdd(&out[(long)erow * 3 + d], coord_diff[(long)e * 3 + d] * scale);
    }
  }
}

__global__ void init_out_kernel(const float* __restrict__ coord,
                                float* __restrict__ out, int n) {
  int i = blockIdx.x * blockDim.x + threadIdx.x;
  if (i < n) out[i] = coord[i];
}

__global__ void finalize_out_kernel(const float* __restrict__ node_mask,
                                    float* __restrict__ out, int n) {
  int i = blockIdx.x * blockDim.x + threadIdx.x;
  if (i < n) out[i] *= node_mask[i / 3];
}

extern "C" void kernel_launch(void* const* d_in, const int* in_sizes, int n_in,
                              void* d_out, int out_size, void* d_ws, size_t ws_size,
                              hipStream_t stream) {
  const float* h          = (const float*)d_in[0];
  const float* coord      = (const float*)d_in[1];
  const int*   edge_index = (const int*)  d_in[2];
  const float* coord_diff = (const float*)d_in[3];
  // d_in[4] = coord_cross (unused by reference path)
  const float* edge_attr  = (const float*)d_in[5];
  const float* node_mask  = (const float*)d_in[6];
  const float* edge_mask  = (const float*)d_in[7];
  const float* W1         = (const float*)d_in[8];
  const float* b1         = (const float*)d_in[9];
  const float* W2         = (const float*)d_in[10];
  const float* b2         = (const float*)d_in[11];
  const float* W3         = (const float*)d_in[12];
  float* out = (float*)d_out;

  const int n_out_elems = NNODES * 3;

  (void)hipFuncSetAttribute((const void*)egnn_edge_kernel,
                            hipFuncAttributeMaxDynamicSharedMemorySize,
                            (int)SMEM_BYTES);

  init_out_kernel<<<(n_out_elems + 255) / 256, 256, 0, stream>>>(coord, out, n_out_elems);

  const int n_wg = NEDGES / EDGES_PER_WG;  // 6250, exact
  egnn_edge_kernel<<<n_wg, THREADS, SMEM_BYTES, stream>>>(
      h, edge_index, coord_diff, edge_attr, edge_mask,
      W1, b1, W2, b2, W3, out);

  finalize_out_kernel<<<(n_out_elems + 255) / 256, 256, 0, stream>>>(node_mask, out, n_out_elems);
}
